// HierConv_59777354826044
// MI455X (gfx1250) — matrix operand order repaired and checked by reading, not compiled, not run on hardware
//
#include <hip/hip_runtime.h>

// MI455X / gfx1250: wave32, WMMA 16x16x32 bf16 (f32 accum).
// f32 accuracy recovered via bf16 hi/lo split: x*w ~= xh*wh + xh*wl + xl*wh.

typedef __bf16 bf16_t;
typedef __attribute__((ext_vector_type(16))) __bf16 v16bf;
typedef __attribute__((ext_vector_type(8)))  float  v8f;

__device__ __forceinline__ v8f wmma_bf16(v16bf a, v16bf b, v8f c) {
  // (neg_a, A, neg_b, B, c_mod, C, reuse_a, reuse_b)
  return __builtin_amdgcn_wmma_f32_16x16x32_bf16(false, a, false, b, (short)0, c, false, false);
}

// ---------------------------------------------------------------------------
// Prep: swizzle weights into WMMA A-fragment layout (bf16 hi/lo), build
// depthwise coefficients and fused bias.
// A-layout (16x32 bf16): lane l -> m = l&15 ; element e -> k = (l>>4)*16 + e
// Fragment storage: idx = ((kt*4 + mt)*32 + l)*16 + e   (kt = tap*2 + icHalf)
// ---------------------------------------------------------------------------
__global__ void k_prep(const float* __restrict__ w3, const float* __restrict__ w5,
                       const float* __restrict__ wr, const float* __restrict__ b5,
                       const float* __restrict__ br,
                       bf16_t* __restrict__ w3hi, bf16_t* __restrict__ w3lo,
                       bf16_t* __restrict__ w5hi, bf16_t* __restrict__ w5lo,
                       float* __restrict__ acoef, float* __restrict__ bias2)
{
  int t = blockIdx.x * blockDim.x + threadIdx.x;
  const int N5 = 50 * 2048, N3 = 18 * 2048;
  if (t < N5) {                                    // 5x5 weights, scaled by wr[c,0]
    int e = t & 15, l = (t >> 4) & 31, mt = (t >> 9) & 3, kt = t >> 11;
    int tap = kt >> 1, q = kt & 1, u = tap / 5, v = tap % 5;
    int oc = mt * 16 + (l & 15);
    int k  = ((l >> 4) << 4) + e;
    int ic = (q << 5) + k;
    float w = wr[oc * 10] * w5[((oc * 64 + ic) * 5 + u) * 5 + v];
    bf16_t h = (bf16_t)w;
    w5hi[t] = h; w5lo[t] = (bf16_t)(w - (float)h);
    return;
  }
  t -= N5;
  if (t < N3) {                                    // 3x3 weights
    int e = t & 15, l = (t >> 4) & 31, mt = (t >> 9) & 3, kt = t >> 11;
    int tap = kt >> 1, q = kt & 1, kh = tap / 3, kw = tap % 3;
    int oc = mt * 16 + (l & 15);
    int k  = ((l >> 4) << 4) + e;
    int ic = (q << 5) + k;
    float w = w3[((oc * 64 + ic) * 3 + kh) * 3 + kw];
    bf16_t h = (bf16_t)w;
    w3hi[t] = h; w3lo[t] = (bf16_t)(w - (float)h);
    return;
  }
  t -= N3;
  if (t < 576) {                                   // depthwise coefs: shifted[h,w]=y3[h+1-sw,w+1-sh]
    int c = t / 9, r9 = t % 9, i = r9 / 3, j = r9 % 3;
    acoef[t] = wr[c * 10 + 1 + (2 - j) * 3 + (2 - i)];
    return;
  }
  t -= 576;
  if (t < 64) bias2[t] = br[t] + wr[t * 10] * b5[t];
}

// ---------------------------------------------------------------------------
// Stage 1: y3 = conv3x3(x,w3)+b3.  Block = (b, h, 64-pixel segment), 4 waves.
// Wave m-tile = 16 OCs; 4 n-tiles of 16 pixels.  K = 9 taps * 64 IC = 18 ktiles.
// LDS: x rows h-1..h+1, cols w0-1..w0+64, layout [row][col][ic] bf16 (hi|lo).
// ---------------------------------------------------------------------------
__global__ __launch_bounds__(128) void k_conv3(
    const float* __restrict__ x, const float* __restrict__ b3,
    const bf16_t* __restrict__ w3hi, const bf16_t* __restrict__ w3lo,
    float* __restrict__ y3)
{
  extern __shared__ char smemRaw[];
  bf16_t* sHi = (bf16_t*)smemRaw;          // 3*66*64 bf16
  bf16_t* sLo = sHi + 3 * 66 * 64;
  const int tid = threadIdx.x, bid = blockIdx.x;
  const int seg = bid & 1, h = (bid >> 1) & 127, b = bid >> 8;
  const int w0 = seg << 6;

  for (int idx = tid; idx < 3 * 66 * 64; idx += 128) {
    int cc = idx % 66, t2 = idx / 66;
    int ic = t2 & 63, r = t2 >> 6;
    int hh = h + r - 1, ww = w0 + cc - 1;
    float vx = 0.f;
    if ((unsigned)hh < 128u && (unsigned)ww < 128u)
      vx = x[(((b << 6) + ic) << 14) + (hh << 7) + ww];
    bf16_t vh = (bf16_t)vx;
    int s = (r * 66 + cc) * 64 + ic;
    sHi[s] = vh;
    sLo[s] = (bf16_t)(vx - (float)vh);
  }
  __syncthreads();

  const int wave = tid >> 5, lane = tid & 31;
  const int nlo = lane & 15, halfB = lane >> 4;
  v8f acc[4];
  for (int i = 0; i < 4; ++i) for (int j = 0; j < 8; ++j) acc[i][j] = 0.f;

  for (int kt = 0; kt < 18; ++kt) {
    int tap = kt >> 1, q = kt & 1;
    int kh = tap / 3, kw = tap % 3;
    const v16bf Ahi = *(const v16bf*)(w3hi + (((kt << 2) + wave) * 32 + lane) * 16);
    const v16bf Alo = *(const v16bf*)(w3lo + (((kt << 2) + wave) * 32 + lane) * 16);
    int icb = (q << 5) + (halfB << 4);   // B-layout: k = (lane>>4)*16 + e
    for (int nt = 0; nt < 4; ++nt) {
      int off = (kh * 66 + (nt << 4) + nlo + kw) * 64 + icb;
      v16bf Bhi = *(const v16bf*)(sHi + off);
      v16bf Blo = *(const v16bf*)(sLo + off);
      acc[nt] = wmma_bf16(Ahi, Bhi, acc[nt]);
      acc[nt] = wmma_bf16(Ahi, Blo, acc[nt]);
      acc[nt] = wmma_bf16(Alo, Bhi, acc[nt]);
    }
  }

  for (int nt = 0; nt < 4; ++nt) {
    int w_ = w0 + (nt << 4) + nlo;
    for (int r = 0; r < 8; ++r) {                  // D: M = r + 8*(lane>>4)
      int oc = (wave << 4) + r + (halfB << 3);
      y3[(((b << 6) + oc) << 14) + (h << 7) + w_] = acc[nt][r] + b3[oc];
    }
  }
}

// ---------------------------------------------------------------------------
// Stage 2: out = conv5x5(x, wr0*w5) + depthwise3x3(y3, acoef) + bias2.
// Block = (b, h, 32-pixel segment); K = 25 taps * 64 IC = 50 ktiles.
// ---------------------------------------------------------------------------
__global__ __launch_bounds__(128) void k_conv5_fuse(
    const float* __restrict__ x,
    const bf16_t* __restrict__ w5hi, const bf16_t* __restrict__ w5lo,
    const float* __restrict__ y3, const float* __restrict__ acoef,
    const float* __restrict__ bias2, float* __restrict__ out)
{
  extern __shared__ char smemRaw[];
  bf16_t* sHi = (bf16_t*)smemRaw;          // 5*36*64 bf16
  bf16_t* sLo = sHi + 5 * 36 * 64;
  const int tid = threadIdx.x, bid = blockIdx.x;
  const int seg = bid & 3, h = (bid >> 2) & 127, b = bid >> 9;
  const int w0 = seg << 5;

  for (int idx = tid; idx < 5 * 36 * 64; idx += 128) {
    int cc = idx % 36, t2 = idx / 36;
    int ic = t2 & 63, r = t2 >> 6;
    int hh = h + r - 2, ww = w0 + cc - 2;
    float vx = 0.f;
    if ((unsigned)hh < 128u && (unsigned)ww < 128u)
      vx = x[(((b << 6) + ic) << 14) + (hh << 7) + ww];
    bf16_t vh = (bf16_t)vx;
    int s = (r * 36 + cc) * 64 + ic;
    sHi[s] = vh;
    sLo[s] = (bf16_t)(vx - (float)vh);
  }
  __syncthreads();

  const int wave = tid >> 5, lane = tid & 31;
  const int nlo = lane & 15, halfB = lane >> 4;
  v8f acc[2];
  for (int i = 0; i < 2; ++i) for (int j = 0; j < 8; ++j) acc[i][j] = 0.f;

  for (int kt = 0; kt < 50; ++kt) {
    int tap = kt >> 1, q = kt & 1;
    int u = tap / 5, v = tap % 5;
    const v16bf Ahi = *(const v16bf*)(w5hi + (((kt << 2) + wave) * 32 + lane) * 16);
    const v16bf Alo = *(const v16bf*)(w5lo + (((kt << 2) + wave) * 32 + lane) * 16);
    int icb = (q << 5) + (halfB << 4);
    for (int nt = 0; nt < 2; ++nt) {
      int off = (u * 36 + (nt << 4) + nlo + v) * 64 + icb;
      v16bf Bhi = *(const v16bf*)(sHi + off);
      v16bf Blo = *(const v16bf*)(sLo + off);
      acc[nt] = wmma_bf16(Ahi, Bhi, acc[nt]);
      acc[nt] = wmma_bf16(Ahi, Blo, acc[nt]);
      acc[nt] = wmma_bf16(Alo, Bhi, acc[nt]);
    }
  }

  for (int nt = 0; nt < 2; ++nt) {
    int w_ = w0 + (nt << 4) + nlo;
    for (int r = 0; r < 8; ++r) {
      int oc = (wave << 4) + r + (halfB << 3);
      float s = bias2[oc];
      const float* y3c = y3 + (size_t)(((b << 6) + oc) << 14);
      for (int i = 0; i < 3; ++i) {
        int hh = h + i - 1;
        if ((unsigned)hh >= 128u) continue;
        for (int j = 0; j < 3; ++j) {
          int ww = w_ + j - 1;
          if ((unsigned)ww < 128u)
            s += acoef[oc * 9 + i * 3 + j] * y3c[(hh << 7) + ww];
        }
      }
      out[(((b << 6) + oc) << 14) + (h << 7) + w_] = acc[nt][r] + s;
    }
  }
}

// ---------------------------------------------------------------------------
extern "C" void kernel_launch(void* const* d_in, const int* in_sizes, int n_in,
                              void* d_out, int out_size, void* d_ws, size_t ws_size,
                              hipStream_t stream)
{
  (void)in_sizes; (void)n_in; (void)out_size; (void)ws_size;
  const float* x  = (const float*)d_in[0];
  const float* w3 = (const float*)d_in[1];
  const float* b3 = (const float*)d_in[2];
  const float* w5 = (const float*)d_in[3];
  const float* b5 = (const float*)d_in[4];
  const float* wr = (const float*)d_in[5];
  const float* br = (const float*)d_in[6];
  float* out = (float*)d_out;
  char* ws = (char*)d_ws;

  // Workspace layout (needs ~34.2 MB): y3 | w3 frags hi/lo | w5 frags hi/lo | acoef | bias2
  const size_t Y3B = 8ull * 64 * 128 * 128 * 4;          // 33,554,432 B
  float*  y3    = (float*)ws;
  bf16_t* w3hi  = (bf16_t*)(ws + Y3B);
  bf16_t* w3lo  = w3hi + 18 * 2048;
  bf16_t* w5hi  = (bf16_t*)(ws + Y3B + 2ull * 18 * 2048 * 2);
  bf16_t* w5lo  = w5hi + 50 * 2048;
  float*  acoef = (float*)(ws + Y3B + 2ull * 18 * 2048 * 2 + 2ull * 50 * 2048 * 2);
  float*  bias2 = acoef + 576;

  const int prepN = 50 * 2048 + 18 * 2048 + 576 + 64;
  k_prep<<<(prepN + 255) / 256, 256, 0, stream>>>(w3, w5, wr, b5, br,
                                                  w3hi, w3lo, w5hi, w5lo, acoef, bias2);
  // (b=8) x (h=128) x (2 segs of 64 px); 50,688 B LDS
  k_conv3<<<2048, 128, 3 * 66 * 64 * 2 * 2, stream>>>(x, b3, w3hi, w3lo, y3);
  // (b=8) x (h=128) x (4 segs of 32 px); 46,080 B LDS
  k_conv5_fuse<<<4096, 128, 5 * 36 * 64 * 2 * 2, stream>>>(x, w5hi, w5lo, y3, acoef, bias2, out);
}
